// Encoder_16681652978153
// MI455X (gfx1250) — matrix operand order repaired and checked by reading, not compiled, not run on hardware
//
#include <hip/hip_runtime.h>
#include <math.h>

// Problem constants (from the reference)
#define TT 256
#define BB 128
#define HH 512
#define EE 512
#define VV 32000
#define KK 50
#define FF 1000
#define G3H (3 * HH)          // 1536
#define TBROWS (TT * BB)      // 32768

typedef __attribute__((ext_vector_type(16))) __bf16 v16bf;
typedef __attribute__((ext_vector_type(8)))  float  v8f;
typedef __attribute__((ext_vector_type(4)))  unsigned int v4u;
typedef __attribute__((ext_vector_type(8)))  int v8i_;
typedef __attribute__((ext_vector_type(4)))  int v4i_;

union Frag16 { v16bf v; v4u q[2]; };

// ---------- small helpers ----------
__device__ inline unsigned short f32_to_bf16(float f) {
  union { float f; unsigned u; } x; x.f = f;
  unsigned r = x.u + 0x7FFFu + ((x.u >> 16) & 1u);   // round-to-nearest-even
  return (unsigned short)(r >> 16);
}
__device__ inline float bf16_to_f32(unsigned short h) {
  union { unsigned u; float f; } x; x.u = ((unsigned)h) << 16;
  return x.f;
}
__device__ inline float sigmoidf_(float x) { return 1.0f / (1.0f + expf(-x)); }

__device__ inline v8f wmma_bf16(v16bf a, v16bf b, v8f c) {
  return __builtin_amdgcn_wmma_f32_16x16x32_bf16(
      /*neg_a=*/false, a, /*neg_b=*/false, b,
      /*c_mod=*/(short)0, c, /*reuse_a=*/false, /*reuse_b=*/false);
}

// low 32 bits of a generic pointer to __shared__ == LDS byte offset
__device__ inline unsigned lds_off(const void* p) { return (unsigned)(size_t)p; }

// ---- Tensor Data Mover: 2-D bf16 tile load (6-arg builtin on this toolchain) ----
__device__ inline void tdm_load_2d_bf16(const unsigned short* gptr, unsigned ldsByteOff,
                                        unsigned tile_d0, unsigned tile_d1,
                                        unsigned tensor_d0, unsigned tensor_d1,
                                        unsigned stride0) {
  unsigned long long ga = (unsigned long long)(size_t)gptr;
  v4u g0;
  g0[0] = 1u;                                              // count=1 (valid), no gather
  g0[1] = ldsByteOff;                                      // lds_addr [63:32]
  g0[2] = (unsigned)(ga & 0xFFFFFFFFu);                    // global_addr lo
  g0[3] = (unsigned)((ga >> 32) & 0x1FFFFFFu) | (2u << 30);// global_addr hi | type=2
  v8i_ g1;
  g1[0] = (int)(1u << 16);                                 // data_size=1 -> 2 bytes
  g1[1] = (int)((tensor_d0 & 0xFFFFu) << 16);              // tensor_dim0[15:0]
  g1[2] = (int)(((tensor_d0 >> 16) & 0xFFFFu) | ((tensor_d1 & 0xFFFFu) << 16));
  g1[3] = (int)(((tensor_d1 >> 16) & 0xFFFFu) | ((tile_d0 & 0xFFFFu) << 16));
  g1[4] = (int)(tile_d1 & 0xFFFFu);                        // tile_dim1 ; tile_dim2=0
  g1[5] = (int)stride0;                                    // tensor_dim0_stride[31:0]
  g1[6] = 0;
  g1[7] = 0;
  v4i_ z4 = {0, 0, 0, 0};
  v8i_ z8 = {0, 0, 0, 0, 0, 0, 0, 0};
  __builtin_amdgcn_tensor_load_to_lds(g0, g1, z4, z4, z8, 0);
}

// A fragment 16x32 bf16 from row-major buffer (lda elems/row)
__device__ inline v16bf load_a_frag(const unsigned short* A, int lda, int rowBase, int k0, int lane) {
  int m  = rowBase + (lane & 15);
  int kb = k0 + ((lane & 16) ? 8 : 0);
  const unsigned short* p = A + (long)m * lda + kb;
  Frag16 f;
  f.q[0] = *(const v4u*)(p);        // K = kb .. kb+7
  f.q[1] = *(const v4u*)(p + 16);   // K = kb+16 .. kb+23
  return f.v;
}

// B fragment 32x16 bf16: B[k][n] = W[n][k], W row-major (ldw elems/row)
__device__ inline v16bf load_b_frag(const unsigned short* W, int ldw, int colBase, int k0, int lane) {
  int n  = colBase + (lane & 15);
  int kb = k0 + ((lane & 16) ? 8 : 0);
  const unsigned short* p = W + (long)n * ldw + kb;
  Frag16 f;
  f.q[0] = *(const v4u*)(p);
  f.q[1] = *(const v4u*)(p + 16);
  return f.v;
}

// software grid barrier (generation-based); bar[0]=count, bar[32]=generation
__device__ inline void grid_sync(unsigned* bar, int nblocks) {
  __syncthreads();
  if (threadIdx.x == 0) {
    unsigned* cnt = bar;
    unsigned* gen = bar + 32;
    __threadfence();
    unsigned g = atomicAdd(gen, 0u);
    unsigned arrived = atomicAdd(cnt, 1u) + 1u;
    if (arrived == (unsigned)nblocks) {
      atomicExch(cnt, 0u);
      __threadfence();
      atomicAdd(gen, 1u);
    } else {
      while (atomicAdd(gen, 0u) == g) { __builtin_amdgcn_s_sleep(1); }
    }
    __threadfence();
  }
  __syncthreads();
}

// ---------- kernels ----------

__global__ void init_kernel(const float* __restrict__ h0,
                            float* hs0, float* hs1,
                            unsigned short* hb0, unsigned short* hb1,
                            unsigned* bar) {
  int i = blockIdx.x * blockDim.x + threadIdx.x;
  if (i < 64) bar[i] = 0u;
  if (i < BB * HH) {
    float a = h0[i], b = h0[BB * HH + i];
    hs0[i] = a; hs1[i] = b;
    hb0[i] = f32_to_bf16(a); hb1[i] = f32_to_bf16(b);
  }
}

__global__ void cvt_bf16_kernel(const float* __restrict__ src, unsigned short* __restrict__ dst, int n) {
  int i = blockIdx.x * blockDim.x + threadIdx.x;
  if (i < n) dst[i] = f32_to_bf16(src[i]);
}

__global__ void embed_kernel(const int* __restrict__ src, const float* __restrict__ table,
                             unsigned short* __restrict__ emb) {
  long row = blockIdx.x;                        // row = t*B + b
  int tok = src[row];
  const float* trow = table + (long)tok * EE;
  unsigned short* erow = emb + row * EE;
  for (int e = threadIdx.x; e < EE; e += blockDim.x) erow[e] = f32_to_bf16(trow[e]);
}

// gx = emb @ w_ih^T + b_ih -> bf16 (M=TBROWS, N=3H, K=E)
// block = 8 waves = 128M x 64N; TDM double-buffers A(128x32) and B(64x32) tiles in LDS.
__global__ void gx_gemm_kernel(const unsigned short* __restrict__ emb,
                               const unsigned short* __restrict__ wih0,
                               const unsigned short* __restrict__ wih1,
                               const float* __restrict__ bih0,
                               const float* __restrict__ bih1,
                               unsigned short* __restrict__ gx0,
                               unsigned short* __restrict__ gx1) {
  __shared__ unsigned short sA[2][128 * 32];   // 8 KB x2
  __shared__ unsigned short sB[2][64 * 32];    // 4 KB x2

  const int dir = blockIdx.z;
  const unsigned short* W = dir ? wih1 : wih0;
  const float* bias       = dir ? bih1 : bih0;
  unsigned short* gx      = dir ? gx1 : gx0;

  const int lane = threadIdx.x & 31;
  const int wid  = threadIdx.x >> 5;              // 0..7
  const int rowBase0 = blockIdx.x * 128;
  const int colBase  = blockIdx.y * 64;

  v8f acc0 = (v8f){0.f,0.f,0.f,0.f,0.f,0.f,0.f,0.f};
  v8f acc1 = acc0, acc2 = acc0, acc3 = acc0;

  const int NK = EE / 32;                          // 16 K-steps

  // prologue: stage K-step 0
  if (wid == 0) {
    tdm_load_2d_bf16(emb + (long)rowBase0 * EE, lds_off(&sA[0][0]), 32, 128, EE, 128, EE);
    tdm_load_2d_bf16(W   + (long)colBase  * EE, lds_off(&sB[0][0]), 32, 64,  EE, 64,  EE);
    __builtin_amdgcn_s_wait_tensorcnt(0);
  }
  __syncthreads();

  for (int kk = 0; kk < NK; ++kk) {
    const int cur = kk & 1;
    // prefetch next K-step into the other buffer while we compute
    if (wid == 0 && (kk + 1) < NK) {
      const int k0n = (kk + 1) * 32;
      tdm_load_2d_bf16(emb + (long)rowBase0 * EE + k0n, lds_off(&sA[cur ^ 1][0]), 32, 128, EE, 128, EE);
      tdm_load_2d_bf16(W   + (long)colBase  * EE + k0n, lds_off(&sB[cur ^ 1][0]), 32, 64,  EE, 64,  EE);
    }
    // batch ALL fragment loads before the first WMMA consumes them
    v16bf a  = load_a_frag(&sA[cur][0], 32, wid * 16, 0, lane);
    v16bf b0 = load_b_frag(&sB[cur][0], 32,  0, 0, lane);
    v16bf b1 = load_b_frag(&sB[cur][0], 32, 16, 0, lane);
    v16bf b2 = load_b_frag(&sB[cur][0], 32, 32, 0, lane);
    v16bf b3 = load_b_frag(&sB[cur][0], 32, 48, 0, lane);
    acc0 = wmma_bf16(a, b0, acc0);
    acc1 = wmma_bf16(a, b1, acc1);
    acc2 = wmma_bf16(a, b2, acc2);
    acc3 = wmma_bf16(a, b3, acc3);
    if (wid == 0) __builtin_amdgcn_s_wait_tensorcnt(0);
    __syncthreads();
  }

  const int mOff = (lane & 16) ? 8 : 0;
  const int nl   = lane & 15;
  const long rB  = rowBase0 + wid * 16 + mOff;
  {
    int col = colBase + nl;        float bv = bias[col];
    #pragma unroll
    for (int r = 0; r < 8; ++r) gx[(rB + r) * (long)G3H + col] = f32_to_bf16(acc0[r] + bv);
  }
  {
    int col = colBase + 16 + nl;   float bv = bias[col];
    #pragma unroll
    for (int r = 0; r < 8; ++r) gx[(rB + r) * (long)G3H + col] = f32_to_bf16(acc1[r] + bv);
  }
  {
    int col = colBase + 32 + nl;   float bv = bias[col];
    #pragma unroll
    for (int r = 0; r < 8; ++r) gx[(rB + r) * (long)G3H + col] = f32_to_bf16(acc2[r] + bv);
  }
  {
    int col = colBase + 48 + nl;   float bv = bias[col];
    #pragma unroll
    for (int r = 0; r < 8; ++r) gx[(rB + r) * (long)G3H + col] = f32_to_bf16(acc3[r] + bv);
  }
}

// persistent GRU scan for one direction. NB == G3H/64 == 24 blocks; block b owns
// w_hh columns [64b, 64b+64) which are TDM-staged into LDS ONCE for all 256 steps.
__global__ void scan_kernel(int dir,
                            const unsigned short* __restrict__ gx,   // (T*B) x 3H bf16
                            const unsigned short* __restrict__ whh,  // 3H x H bf16
                            const float* __restrict__ bhh,           // 3H
                            const int* __restrict__ lengths,         // B
                            float* __restrict__ hstate,              // B x H
                            unsigned short* __restrict__ hbf,        // B x H bf16
                            float* __restrict__ gh,                  // B x 3H
                            float* __restrict__ outbuf,              // d_out output section
                            unsigned* bar) {
  __shared__ unsigned short sW[64 * HH];          // 64 cols x 512 K bf16 = 64 KB

  const int NB   = gridDim.x;                     // 24
  const int lane = threadIdx.x & 31;
  const int wid  = threadIdx.x >> 5;              // 0..7 -> mtile
  const int gtid = blockIdx.x * blockDim.x + threadIdx.x;
  const int nthreads = NB * blockDim.x;
  const int colBase  = blockIdx.x * 64;

  // one-time weight panel stage (reused for all T steps)
  if (wid == 0) {
    tdm_load_2d_bf16(whh + (long)colBase * HH, lds_off(&sW[0]), HH, 64, HH, 64, HH);
    __builtin_amdgcn_s_wait_tensorcnt(0);
  }
  __syncthreads();

  for (int step = 0; step < TT; ++step) {
    const int t = dir ? (TT - 1 - step) : step;

    // ---- phase 1: gh = h @ w_hh^T  (each wave: one 16M x 64N tile) ----
    {
      const int rowBase = wid * 16;
      v8f acc0 = (v8f){0.f,0.f,0.f,0.f,0.f,0.f,0.f,0.f};
      v8f acc1 = acc0, acc2 = acc0, acc3 = acc0;
      #pragma unroll 4
      for (int k0 = 0; k0 < HH; k0 += 32) {
        v16bf a  = load_a_frag(hbf, HH, rowBase, k0, lane);
        v16bf b0 = load_b_frag(&sW[0], HH,  0, k0, lane);   // LDS-resident weights
        v16bf b1 = load_b_frag(&sW[0], HH, 16, k0, lane);
        v16bf b2 = load_b_frag(&sW[0], HH, 32, k0, lane);
        v16bf b3 = load_b_frag(&sW[0], HH, 48, k0, lane);
        acc0 = wmma_bf16(a, b0, acc0);
        acc1 = wmma_bf16(a, b1, acc1);
        acc2 = wmma_bf16(a, b2, acc2);
        acc3 = wmma_bf16(a, b3, acc3);
      }
      const int mOff = (lane & 16) ? 8 : 0;
      const int nl   = lane & 15;
      const int rB   = rowBase + mOff;
      #pragma unroll
      for (int r = 0; r < 8; ++r) gh[(rB + r) * G3H + colBase      + nl] = acc0[r];
      #pragma unroll
      for (int r = 0; r < 8; ++r) gh[(rB + r) * G3H + colBase + 16 + nl] = acc1[r];
      #pragma unroll
      for (int r = 0; r < 8; ++r) gh[(rB + r) * G3H + colBase + 32 + nl] = acc2[r];
      #pragma unroll
      for (int r = 0; r < 8; ++r) gh[(rB + r) * G3H + colBase + 48 + nl] = acc3[r];
    }
    grid_sync(bar, NB);

    // ---- phase 2: gate update ----
    const unsigned short* gxt = gx + (long)t * BB * G3H;
    for (int i = gtid; i < BB * HH; i += nthreads) {
      int b = i >> 9;
      int n = i & (HH - 1);
      const float* ghrow = gh + b * G3H;
      float ghr = ghrow[n]        + bhh[n];
      float ghz = ghrow[HH + n]   + bhh[HH + n];
      float ghn = ghrow[2*HH + n] + bhh[2*HH + n];
      const unsigned short* gxrow = gxt + (long)b * G3H;
      float gxr = bf16_to_f32(gxrow[n]);
      float gxz = bf16_to_f32(gxrow[HH + n]);
      float gxn = bf16_to_f32(gxrow[2*HH + n]);
      float r  = sigmoidf_(gxr + ghr);
      float z  = sigmoidf_(gxz + ghz);
      float nn = tanhf(gxn + r * ghn);
      float hold = hstate[i];
      float hnew = (1.0f - z) * nn + z * hold;
      bool  m = (t < lengths[b]);
      float h = m ? hnew : hold;
      hstate[i] = h;
      hbf[i] = f32_to_bf16(h);
      long oidx = (long)t * (BB * HH) + i;
      if (dir == 0) {
        outbuf[oidx] = h;                              // raw hs_f
      } else {
        float hf = outbuf[oidx];
        outbuf[oidx] = m ? (hf + h) : 0.0f;            // (hs_f + hs_b) * mask
      }
    }
    grid_sync(bar, NB);
  }
}

__global__ void hidden_kernel(const float* __restrict__ hs0, const float* __restrict__ hs1,
                              float* __restrict__ out_hidden) {
  int i = blockIdx.x * blockDim.x + threadIdx.x;
  if (i < BB * HH) {
    out_hidden[i]           = tanhf(hs0[i]);
    out_hidden[BB * HH + i] = tanhf(hs1[i]);
  }
}

__global__ void codes_kernel(const int* __restrict__ src, const int* __restrict__ nmf,
                             const float* __restrict__ topics, float* __restrict__ codes_out) {
  __shared__ float flags[FF];
  int b = blockIdx.x;
  for (int f = threadIdx.x; f < FF; f += blockDim.x) flags[f] = 0.0f;
  __syncthreads();
  for (int t = threadIdx.x; t < TT; t += blockDim.x) {
    int idx = nmf[src[(long)t * BB + b]];
    if (idx >= 0) flags[idx] = 1.0f;
  }
  __syncthreads();
  for (int k = threadIdx.x; k < KK; k += blockDim.x) {
    float s = 0.0f;
    for (int f = 0; f < FF; ++f) s += topics[k * FF + f] * flags[f];
    codes_out[b * KK + k] = s;
  }
}

// ---------- launch ----------
extern "C" void kernel_launch(void* const* d_in, const int* in_sizes, int n_in,
                              void* d_out, int out_size, void* d_ws, size_t ws_size,
                              hipStream_t stream) {
  (void)in_sizes; (void)n_in; (void)out_size; (void)ws_size;

  const int*   src     = (const int*)  d_in[0];
  const int*   lens    = (const int*)  d_in[1];
  const int*   nmf     = (const int*)  d_in[2];
  const float* table   = (const float*)d_in[3];
  const float* topics  = (const float*)d_in[4];
  const float* w_ih_f  = (const float*)d_in[5];
  const float* w_hh_f  = (const float*)d_in[6];
  const float* b_ih_f  = (const float*)d_in[7];
  const float* b_hh_f  = (const float*)d_in[8];
  const float* w_ih_b  = (const float*)d_in[9];
  const float* w_hh_b  = (const float*)d_in[10];
  const float* b_ih_b  = (const float*)d_in[11];
  const float* b_hh_b  = (const float*)d_in[12];
  const float* h0      = (const float*)d_in[13];

  float* out_output = (float*)d_out;                               // T*B*H
  float* out_hidden = out_output + (long)TT * BB * HH;             // 2*B*H
  float* out_codes  = out_hidden + 2L * BB * HH;                   // B*K

  char* w = (char*)d_ws;
  size_t off = 0;
  auto take = [&](size_t bytes) -> void* {
    void* p = w + off;
    off = (off + bytes + 255) & ~(size_t)255;
    return p;
  };
  unsigned*       bar    = (unsigned*)      take(256);
  unsigned short* emb    = (unsigned short*)take((size_t)TBROWS * EE * 2);
  unsigned short* wihbf0 = (unsigned short*)take((size_t)G3H * EE * 2);
  unsigned short* wihbf1 = (unsigned short*)take((size_t)G3H * EE * 2);
  unsigned short* whhbf0 = (unsigned short*)take((size_t)G3H * HH * 2);
  unsigned short* whhbf1 = (unsigned short*)take((size_t)G3H * HH * 2);
  unsigned short* gx0    = (unsigned short*)take((size_t)TBROWS * G3H * 2);
  unsigned short* gx1    = (unsigned short*)take((size_t)TBROWS * G3H * 2);
  float*          gh     = (float*)         take((size_t)BB * G3H * 4);
  float*          hs0    = (float*)         take((size_t)BB * HH * 4);
  float*          hs1    = (float*)         take((size_t)BB * HH * 4);
  unsigned short* hb0    = (unsigned short*)take((size_t)BB * HH * 2);
  unsigned short* hb1    = (unsigned short*)take((size_t)BB * HH * 2);

  init_kernel<<<(BB * HH + 255) / 256, 256, 0, stream>>>(h0, hs0, hs1, hb0, hb1, bar);

  const int wihN = G3H * EE, whhN = G3H * HH;
  cvt_bf16_kernel<<<(wihN + 255) / 256, 256, 0, stream>>>(w_ih_f, wihbf0, wihN);
  cvt_bf16_kernel<<<(wihN + 255) / 256, 256, 0, stream>>>(w_ih_b, wihbf1, wihN);
  cvt_bf16_kernel<<<(whhN + 255) / 256, 256, 0, stream>>>(w_hh_f, whhbf0, whhN);
  cvt_bf16_kernel<<<(whhN + 255) / 256, 256, 0, stream>>>(w_hh_b, whhbf1, whhN);

  embed_kernel<<<TBROWS, 256, 0, stream>>>(src, table, emb);

  dim3 ggrid(TBROWS / 128, G3H / 64, 2);
  gx_gemm_kernel<<<ggrid, 256, 0, stream>>>(emb, wihbf0, wihbf1, b_ih_f, b_ih_b, gx0, gx1);

  const int NB = G3H / 64;   // 24 blocks; block b owns w_hh columns [64b, 64b+64)
  scan_kernel<<<NB, 256, 0, stream>>>(0, gx0, whhbf0, b_hh_f, lens, hs0, hb0, gh, out_output, bar);
  scan_kernel<<<NB, 256, 0, stream>>>(1, gx1, whhbf1, b_hh_b, lens, hs1, hb1, gh, out_output, bar);

  hidden_kernel<<<(BB * HH + 255) / 256, 256, 0, stream>>>(hs0, hs1, out_hidden);
  codes_kernel<<<BB, 256, 0, stream>>>(src, nmf, topics, out_codes);
}
